// LSTMReset_82136954568797
// MI455X (gfx1250) — matrix operand order repaired and checked by reading, not compiled, not run on hardware
//
#include <hip/hip_runtime.h>

typedef __bf16 bf16_t;
typedef __attribute__((ext_vector_type(16))) __bf16 v16bf;
typedef __attribute__((ext_vector_type(8)))  float  v8f;

constexpr int Tn   = 128;   // timesteps
constexpr int Bn   = 512;   // batch
constexpr int Hn   = 512;   // hidden (== IN)
constexpr int G4H  = 2048;  // 4*H
constexpr int Ln   = 2;     // layers
constexpr int Kdim = 1024;  // x(512) + h(512)

// block tile: 64 rows x 32 h-cols (x4 gates), 8 waves of 32
constexpr int MB   = 64;
constexpr int NB   = 32;
constexpr int KC   = 32;    // K chunk per WMMA
constexpr int NKC  = Kdim / KC;
constexpr int ASTR = 40;    // LDS row stride in bf16 elems (80B, 16B-aligned, bank-spread)
constexpr int BSTR = 40;

union U16 { v16bf v; uint4 u[2]; };

__device__ __forceinline__ float fsigmoid(float x) {
  return 1.0f / (1.0f + __expf(-x));
}
__device__ __forceinline__ float ftanh(float x) {
  x = fminf(15.0f, fmaxf(-15.0f, x));
  float e = __expf(2.0f * x);
  return (e - 1.0f) / (e + 1.0f);
}

// One LSTM cell step for one layer: gates = x@Wih^T + h@Whh^T + bsum, then
// elementwise update. Each wave owns one 16x16 (row, h-col) tile across all
// four gates simultaneously so the epilogue is lane-local. LDS tiles are
// double-buffered; weight tiles stream in via global_load_async_to_lds_b128.
__global__ __launch_bounds__(256)
void lstm_step_kernel(const float* __restrict__ x,        // Bn x 512
                      const float* __restrict__ h_prev,   // Bn x 512
                      float*       __restrict__ c_state,  // Bn x 512 (in place)
                      float*       __restrict__ h_out,    // Bn x 512
                      float*       __restrict__ hidden_out,// Bn x 512 or null
                      const bf16_t* __restrict__ Wih,     // 2048 x 512 bf16
                      const bf16_t* __restrict__ Whh,     // 2048 x 512 bf16
                      const float* __restrict__ bsum,     // 2048
                      const int*   __restrict__ reset_t)  // Bn
{
  __shared__ bf16_t As[2][MB * ASTR];        // 64 x 32 activations (bf16)
  __shared__ bf16_t Bs[2][4 * NB * BSTR];    // 128 x 32 weight rows (bf16)

  const int tid   = threadIdx.x;
  const int lane  = tid & 31;
  const int wave  = tid >> 5;     // 0..7
  const int mIdx  = wave & 3;     // 16-row tile within block
  const int nHalf = wave >> 2;    // 16-col half within block
  const int mBase = blockIdx.y * MB;
  const int nBase = blockIdx.x * NB;

  // staging coordinates
  const int aRow  = tid >> 2;     // 0..63, 8 floats each
  const int aQ    = tid & 3;
  const int bRow  = tid >> 1;     // 0..127, 16 bf16 each
  const int bHalf = tid & 1;
  const int bGate = bRow >> 5;
  const int bN    = bRow & 31;
  const float hmask = (reset_t[mBase + aRow] != 0) ? 0.0f : 1.0f;
  const bf16_t* wRowIh = Wih + (size_t)(bGate * Hn + nBase + bN) * Hn + bHalf * 16;
  const bf16_t* wRowHh = Whh + (size_t)(bGate * Hn + nBase + bN) * Hn + bHalf * 16;

  // stage the 64x32 fp32 activation tile -> bf16 LDS (mask on h-part rows)
  auto stageA = [&](int kc, bf16_t* dst) {
    const int  k0  = kc * KC;
    const bool isX = (k0 < Hn);
    const int  kk  = isX ? k0 : (k0 - Hn);
    const float* src = (isX ? x : h_prev) + (size_t)(mBase + aRow) * Hn + kk + aQ * 8;
    const float4 v0 = *reinterpret_cast<const float4*>(src);
    const float4 v1 = *reinterpret_cast<const float4*>(src + 4);
    const float  s  = isX ? 1.0f : hmask;
    union { uint4 u; bf16_t b[8]; } pk;
    pk.b[0] = (bf16_t)(v0.x * s); pk.b[1] = (bf16_t)(v0.y * s);
    pk.b[2] = (bf16_t)(v0.z * s); pk.b[3] = (bf16_t)(v0.w * s);
    pk.b[4] = (bf16_t)(v1.x * s); pk.b[5] = (bf16_t)(v1.y * s);
    pk.b[6] = (bf16_t)(v1.z * s); pk.b[7] = (bf16_t)(v1.w * s);
    *reinterpret_cast<uint4*>(dst + aRow * ASTR + aQ * 8) = pk.u;
    if (kc + 2 < NKC) {   // prefetch next-next activation chunk
      const int  k2  = (kc + 2) * KC;
      const bool iX2 = (k2 < Hn);
      const float* ps = (iX2 ? x : h_prev) + (size_t)(mBase + aRow) * Hn +
                        (iX2 ? k2 : k2 - Hn) + aQ * 8;
      __builtin_prefetch(ps, 0, 0);        // global_prefetch_b8
    }
  };

  // stage 128 weight rows x 32 k (already bf16) straight into LDS, async.
  // ISA: INST_OFFSET is added to both the LDS and the global address, so one
  // base pair covers both 16B halves. Tracked on ASYNCcnt.
  auto stageB = [&](int kc, bf16_t* dst) {
    const int  k0  = kc * KC;
    const bool isX = (k0 < Hn);
    const int  kk  = isX ? k0 : (k0 - Hn);
    const bf16_t* wsrc = (isX ? wRowIh : wRowHh) + kk;
    const uint32_t l0 = (uint32_t)(uintptr_t)(dst + bRow * BSTR + bHalf * 16);
    const uint64_t g0 = (uint64_t)(uintptr_t)wsrc;
    asm volatile("global_load_async_to_lds_b128 %0, %1, off\n\t"
                 "global_load_async_to_lds_b128 %0, %1, off offset:16"
                 :: "v"(l0), "v"(g0) : "memory");
  };

  v8f zero = {};
  v8f acc[4];
  acc[0] = zero; acc[1] = zero; acc[2] = zero; acc[3] = zero;

  // ---- pipeline prologue ----
  stageA(0, As[0]);
  stageB(0, Bs[0]);
  asm volatile("s_wait_asynccnt 0" ::: "memory");
  __syncthreads();

  #pragma unroll 1
  for (int kc = 0; kc < NKC; ++kc) {
    const int cur = kc & 1;
    const int nxt = cur ^ 1;

    // ---- A fragment: 16x32 bf16, ISA layout (lane<16: M, lane>=16 upper K half) ----
    U16 a;
    {
      const int m = lane & 15, kh = lane >> 4;
      const bf16_t* ap = &As[cur][(mIdx * 16 + m) * ASTR + kh * 8];
      a.u[0] = *reinterpret_cast<const uint4*>(ap);        // K = kh*8 + 0..7
      a.u[1] = *reinterpret_cast<const uint4*>(ap + 16);   // K = 16 + kh*8 + 0..7
    }
    // ---- B fragments for the 4 gates ----
    U16 bfr[4];
    #pragma unroll
    for (int g = 0; g < 4; ++g) {
      const int n = lane & 15, kh = lane >> 4;
      const bf16_t* bp = &Bs[cur][(g * 32 + nHalf * 16 + n) * BSTR + kh * 8];
      bfr[g].u[0] = *reinterpret_cast<const uint4*>(bp);      // K = kh*8 + 0..7
      bfr[g].u[1] = *reinterpret_cast<const uint4*>(bp + 16); // K = 16 + kh*8 + 0..7
    }
    // ---- stage next chunk into the other buffer while WMMAs run ----
    if (kc + 1 < NKC) {
      stageA(kc + 1, As[nxt]);
      stageB(kc + 1, Bs[nxt]);
    }
    // ---- 4 gate WMMAs sharing the A fragment ----
    #pragma unroll
    for (int g = 0; g < 4; ++g) {
      acc[g] = __builtin_amdgcn_wmma_f32_16x16x32_bf16(
          false, a.v, false, bfr[g].v, (short)0, acc[g], false, false);
    }
    asm volatile("s_wait_asynccnt 0" ::: "memory");
    __syncthreads();
  }

  // ---- lane-local LSTM cell epilogue ----
  const int nCol = nBase + nHalf * 16 + (lane & 15);
  const float bi = bsum[nCol];
  const float bf = bsum[Hn + nCol];
  const float bg = bsum[2 * Hn + nCol];
  const float bo = bsum[3 * Hn + nCol];
  const int rBase = mBase + mIdx * 16 + ((lane >> 4) << 3);
  #pragma unroll
  for (int r = 0; r < 8; ++r) {
    const int row = rBase + r;
    const int off = row * Hn + nCol;
    const float cp = (reset_t[row] != 0) ? 0.0f : c_state[off];
    const float gi = fsigmoid(acc[0][r] + bi);
    const float gf = fsigmoid(acc[1][r] + bf);
    const float gg = ftanh  (acc[2][r] + bg);
    const float go = fsigmoid(acc[3][r] + bo);
    const float cn = gf * cp + gi * gg;
    const float hn = go * ftanh(cn);
    c_state[off] = cn;
    h_out[off]   = hn;
    if (hidden_out) hidden_out[off] = hn;
  }
}

__global__ void cvt_f32_bf16_kernel(const float* __restrict__ in,
                                    bf16_t* __restrict__ out, int n) {
  int i = blockIdx.x * blockDim.x + threadIdx.x;
  if (i < n) out[i] = (bf16_t)in[i];
}

__global__ void bias_sum_kernel(const float* __restrict__ a,
                                const float* __restrict__ b,
                                float* __restrict__ o, int n) {
  int i = blockIdx.x * blockDim.x + threadIdx.x;
  if (i < n) o[i] = a[i] + b[i];
}

__global__ void copy_f32_kernel(const float* __restrict__ in,
                                float* __restrict__ out, int n) {
  int i = blockIdx.x * blockDim.x + threadIdx.x;
  if (i < n) out[i] = in[i];
}

extern "C" void kernel_launch(void* const* d_in, const int* in_sizes, int n_in,
                              void* d_out, int out_size, void* d_ws, size_t ws_size,
                              hipStream_t stream) {
  const float* latent = (const float*)d_in[0];
  const float* h0     = (const float*)d_in[1];
  const float* c0     = (const float*)d_in[2];
  const int*   reset  = (const int*)  d_in[3];
  const float* Wih0   = (const float*)d_in[4];
  const float* Whh0   = (const float*)d_in[5];
  const float* bih0   = (const float*)d_in[6];
  const float* bhh0   = (const float*)d_in[7];
  const float* Wih1   = (const float*)d_in[8];
  const float* Whh1   = (const float*)d_in[9];
  const float* bih1   = (const float*)d_in[10];
  const float* bhh1   = (const float*)d_in[11];

  float* out_hidden = (float*)d_out;
  float* out_hn     = out_hidden + (size_t)Tn * Bn * Hn;
  float* out_cn     = out_hn + (size_t)Ln * Bn * Hn;

  // carve workspace (~14 MB), 256B-aligned slices
  char* p = (char*)d_ws;
  auto take = [&](size_t bytes) -> char* {
    char* r = p; p += (bytes + 255) & ~(size_t)255; return r;
  };
  const size_t wElems = (size_t)G4H * Hn;     // 1M per weight matrix
  const size_t sElems = (size_t)Bn * Hn;      // 256K per state buffer
  bf16_t* wih0b = (bf16_t*)take(wElems * 2);
  bf16_t* whh0b = (bf16_t*)take(wElems * 2);
  bf16_t* wih1b = (bf16_t*)take(wElems * 2);
  bf16_t* whh1b = (bf16_t*)take(wElems * 2);
  float*  bsum0 = (float*)take(G4H * 4);
  float*  bsum1 = (float*)take(G4H * 4);
  float*  h0a   = (float*)take(sElems * 4);
  float*  h0b   = (float*)take(sElems * 4);
  float*  h1a   = (float*)take(sElems * 4);
  float*  h1b   = (float*)take(sElems * 4);
  float*  cst0  = (float*)take(sElems * 4);
  float*  cst1  = (float*)take(sElems * 4);

  const int TPB = 256;
  const int wBlocks = (int)((wElems + TPB - 1) / TPB);
  const int sBlocks = (int)((sElems + TPB - 1) / TPB);
  const int gBlocks = (G4H + TPB - 1) / TPB;

  // weights -> bf16 once per call; bias sums; initial states
  cvt_f32_bf16_kernel<<<wBlocks, TPB, 0, stream>>>(Wih0, wih0b, (int)wElems);
  cvt_f32_bf16_kernel<<<wBlocks, TPB, 0, stream>>>(Whh0, whh0b, (int)wElems);
  cvt_f32_bf16_kernel<<<wBlocks, TPB, 0, stream>>>(Wih1, wih1b, (int)wElems);
  cvt_f32_bf16_kernel<<<wBlocks, TPB, 0, stream>>>(Whh1, whh1b, (int)wElems);
  bias_sum_kernel<<<gBlocks, TPB, 0, stream>>>(bih0, bhh0, bsum0, G4H);
  bias_sum_kernel<<<gBlocks, TPB, 0, stream>>>(bih1, bhh1, bsum1, G4H);
  copy_f32_kernel<<<sBlocks, TPB, 0, stream>>>(h0,          h0a, (int)sElems);
  copy_f32_kernel<<<sBlocks, TPB, 0, stream>>>(h0 + sElems, h1a, (int)sElems);
  copy_f32_kernel<<<sBlocks, TPB, 0, stream>>>(c0,          cst0, (int)sElems);
  copy_f32_kernel<<<sBlocks, TPB, 0, stream>>>(c0 + sElems, cst1, (int)sElems);

  // recurrence: per-step, per-layer GEMM+cell; h ping-pongs, c in place
  dim3 grid(Hn / NB, Bn / MB);   // (16, 8)
  for (int t = 0; t < Tn; ++t) {
    const float* xp  = latent + (size_t)t * Bn * Hn;
    const int*   rst = reset + (size_t)t * Bn;
    float* hp0 = (t & 1) ? h0b : h0a;
    float* hn0 = (t & 1) ? h0a : h0b;
    float* hp1 = (t & 1) ? h1b : h1a;
    float* hn1 = (t & 1) ? h1a : h1b;
    lstm_step_kernel<<<grid, TPB, 0, stream>>>(
        xp, hp0, cst0, hn0, nullptr, wih0b, whh0b, bsum0, rst);
    lstm_step_kernel<<<grid, TPB, 0, stream>>>(
        hn0, hp1, cst1, hn1, out_hidden + (size_t)t * Bn * Hn,
        wih1b, whh1b, bsum1, rst);
  }

  // Tn even -> final h states live in the "a" buffers
  copy_f32_kernel<<<sBlocks, TPB, 0, stream>>>(h0a,  out_hn,           (int)sElems);
  copy_f32_kernel<<<sBlocks, TPB, 0, stream>>>(h1a,  out_hn + sElems,  (int)sElems);
  copy_f32_kernel<<<sBlocks, TPB, 0, stream>>>(cst0, out_cn,           (int)sElems);
  copy_f32_kernel<<<sBlocks, TPB, 0, stream>>>(cst1, out_cn + sElems,  (int)sElems);
}
